// attention_head_36386962932292
// MI455X (gfx1250) — matrix operand order repaired and checked by reading, not compiled, not run on hardware
//
#include <hip/hip_runtime.h>

// Problem constants (from reference): B=8, T=2048, C=1024, H=64
#define B_  8
#define T_  2048
#define C_  1024
#define H_  64
#define NT_TILES (T_/16)        // 128
#define SCALE 0.125f            // 64^-0.5

typedef __bf16 bf16_t;
typedef bf16_t v16bf __attribute__((ext_vector_type(16)));
typedef bf16_t v8bf  __attribute__((ext_vector_type(8)));
typedef float  v8f   __attribute__((ext_vector_type(8)));
typedef unsigned int u32x4 __attribute__((ext_vector_type(4)));
typedef int          i32x4 __attribute__((ext_vector_type(4)));
typedef int          i32x8 __attribute__((ext_vector_type(8)));

#if __has_builtin(__builtin_amdgcn_tensor_load_to_lds) && \
    __has_builtin(__builtin_amdgcn_s_wait_tensorcnt)
#define USE_TDM 1
#else
#define USE_TDM 0
#endif

__device__ __forceinline__ unsigned short f32_to_bf16_bits(float f) {
    union { bf16_t b; unsigned short u; } c;
    c.b = (bf16_t)f;                         // native v_cvt, RNE
    return c.u;
}

__device__ __forceinline__ v16bf cat8(v8bf lo, v8bf hi) {
    return __builtin_shufflevector(lo, hi, 0, 1, 2, 3, 4, 5, 6, 7,
                                           8, 9, 10, 11, 12, 13, 14, 15);
}

// A-frag (16x32 bf16) from a row-major bf16 row: lane<16 takes K in
// {off..off+15}, lane>=16 takes {off+16..off+31} (contiguous, matches B).
__device__ __forceinline__ v16bf load_a_frag(const unsigned short* row,
                                             int lhi, int off) {
    v8bf lo = *(const v8bf*)(row + off + lhi * 16);
    v8bf hi = *(const v8bf*)(row + off + lhi * 16 + 8);
    return cat8(lo, hi);
}

// B-frag (32x16 bf16): lane n holds column n, K = off + lhi*16 + j ->
// one contiguous 32B chunk of the source row.
__device__ __forceinline__ v16bf load_b_frag(const unsigned short* row,
                                             int lhi, int off) {
    return *(const v16bf*)(row + off + lhi * 16);
}

#define WMMA_BF16(A, Bm, Cacc) \
    __builtin_amdgcn_wmma_f32_16x16x32_bf16(false, (A), false, (Bm), (short)0, (Cacc), false, false)

// -------------------------------------------------------------------------
// Kernel 0: elementwise fp32 -> bf16 (8 elems/thread, 16B vector stores).
// -------------------------------------------------------------------------
__global__ __launch_bounds__(256)
void cvt_bf16_kernel(const float* __restrict__ src,
                     unsigned short* __restrict__ dst, int n8)
{
    const int i = blockIdx.x * 256 + threadIdx.x;
    if (i >= n8) return;
    const float4 f0 = ((const float4*)src)[i * 2 + 0];
    const float4 f1 = ((const float4*)src)[i * 2 + 1];
    v8bf o;
    o[0] = (bf16_t)f0.x; o[1] = (bf16_t)f0.y;
    o[2] = (bf16_t)f0.z; o[3] = (bf16_t)f0.w;
    o[4] = (bf16_t)f1.x; o[5] = (bf16_t)f1.y;
    o[6] = (bf16_t)f1.z; o[7] = (bf16_t)f1.w;
    ((v8bf*)dst)[i] = o;
}

// -------------------------------------------------------------------------
// Kernel 1: q/k/v = x @ W^T in bf16 fragments, fp32 accumulate.
// The 16x1024 bf16 x-tile (32 KB) is staged into LDS by one Tensor Data
// Mover DMA (when the toolchain exposes the builtin); the wave's vmem pipe
// then exclusively streams W fragments (double-buffered) while A-frags come
// from LDS.  Fallback: direct global A-frag loads.
// -------------------------------------------------------------------------
__global__ __launch_bounds__(32)
void qkv_proj_kernel(const unsigned short* __restrict__ xb,
                     const unsigned short* __restrict__ Wqb,
                     const unsigned short* __restrict__ Wkb,
                     const unsigned short* __restrict__ Wvb,
                     unsigned short* __restrict__ qb,
                     unsigned short* __restrict__ kb,
                     unsigned short* __restrict__ vt)
{
    const int lane  = threadIdx.x;
    const int lrow  = lane & 15;
    const int lhi   = lane >> 4;
    const int which = blockIdx.y;                 // 0=q 1=k 2=v
    const int b     = blockIdx.x / NT_TILES;
    const int t0    = (blockIdx.x % NT_TILES) * 16;

    const unsigned short* W = (which == 0) ? Wqb : (which == 1) ? Wkb : Wvb;
    const unsigned short* wrow = W + (size_t)lrow * C_;

#if USE_TDM
    __shared__ unsigned short xtile[16 * C_];     // 32 KB tile stage
    {
        const unsigned short* gsrc = xb + ((size_t)b * T_ + t0) * C_;
        const unsigned long long ga = (unsigned long long)(size_t)gsrc;
        const unsigned lds_addr = (unsigned)(size_t)(void*)&xtile[0];

        // D# group 0: count=1 | lds_addr | global_addr[56:0] | type=2
        u32x4 g0;
        g0[0] = 1u;
        g0[1] = lds_addr;
        g0[2] = (unsigned)(ga & 0xFFFFFFFFu);
        g0[3] = (unsigned)((ga >> 32) & 0x01FFFFFFu) | (2u << 30);

        // D# group 1: data_size=2B; tensor 1024x16; tile 1024x16; stride 1024
        i32x8 g1;
        g1[0] = (int)(1u << 16);                       // data_size = 2 bytes
        g1[1] = (int)(((unsigned)C_ & 0xFFFFu) << 16); // tensor_dim0[15:0]@48
        g1[2] = (int)((((unsigned)C_ >> 16) & 0xFFFFu) // tensor_dim0[31:16]
                      | (16u << 16));                  // tensor_dim1[15:0]
        g1[3] = (int)((unsigned)C_ << 16);             // tile_dim0 @ [127:112]
        g1[4] = 16;                                    // tile_dim1 (tile_dim2=0)
        g1[5] = (int)C_;                               // tensor_dim0_stride lo
        g1[6] = 0;
        g1[7] = 0;

        const i32x4 gz = { 0, 0, 0, 0 };
#if __clang_major__ >= 23
        const i32x8 gz8 = { 0, 0, 0, 0, 0, 0, 0, 0 };
        __builtin_amdgcn_tensor_load_to_lds(g0, g1, gz, gz, gz8, 0);
#else
        __builtin_amdgcn_tensor_load_to_lds(g0, g1, gz, gz, 0);
#endif
        __builtin_amdgcn_s_wait_tensorcnt(0);
    }
    const unsigned short* arow = &xtile[(size_t)lrow * C_];
#else
    const unsigned short* arow = xb + ((size_t)b * T_ + (t0 + lrow)) * C_;
#endif

    v8f acc[4] = { {}, {}, {}, {} };              // 4 n-tiles cover H=64

    // prime buffer 0
    v16bf a0 = load_a_frag(arow, lhi, 0);
    v16bf b0[4];
    #pragma unroll
    for (int nt = 0; nt < 4; ++nt)
        b0[nt] = load_b_frag(wrow + (size_t)nt * 16 * C_, lhi, 0);

    for (int c0 = 0; c0 < C_; c0 += 64) {
        // fetch buffer 1 (c0+32) while buffer 0 computes
        v16bf a1 = load_a_frag(arow, lhi, c0 + 32);
        v16bf b1[4];
        #pragma unroll
        for (int nt = 0; nt < 4; ++nt)
            b1[nt] = load_b_frag(wrow + (size_t)nt * 16 * C_, lhi, c0 + 32);
#if !USE_TDM
        __builtin_prefetch(arow + c0 + 64, 0, 3);
#endif

        #pragma unroll
        for (int nt = 0; nt < 4; ++nt)
            acc[nt] = WMMA_BF16(a0, b0[nt], acc[nt]);

        // fetch buffer 0 for next iteration (c0+64) under buffer-1 WMMAs
        if (c0 + 64 < C_) {
            a0 = load_a_frag(arow, lhi, c0 + 64);
            #pragma unroll
            for (int nt = 0; nt < 4; ++nt)
                b0[nt] = load_b_frag(wrow + (size_t)nt * 16 * C_, lhi, c0 + 64);
        }

        #pragma unroll
        for (int nt = 0; nt < 4; ++nt)
            acc[nt] = WMMA_BF16(a1, b1[nt], acc[nt]);
    }

    if (which == 2) {
        // vt[b][h][t]: per lane 8 consecutive t -> one 16B store per n-tile
        #pragma unroll
        for (int nt = 0; nt < 4; ++nt) {
            v8bf o;
            #pragma unroll
            for (int r = 0; r < 8; ++r) o[r] = (bf16_t)acc[nt][r];
            *(v8bf*)(vt + ((size_t)b * H_ + nt * 16 + lrow) * T_ + t0 + lhi * 8) = o;
        }
    } else {
        unsigned short* out = (which == 0) ? qb : kb;
        #pragma unroll
        for (int nt = 0; nt < 4; ++nt) {
            #pragma unroll
            for (int r = 0; r < 8; ++r) {
                const int t = t0 + r + lhi * 8;                 // C-layout row
                out[((size_t)b * T_ + t) * H_ + nt * 16 + lrow] =
                    f32_to_bf16_bits(acc[nt][r]);
            }
        }
    }
}

// -------------------------------------------------------------------------
// Kernel 2: per-key-column softmax stats (softmax over the QUERY axis:
// column s normalized over t in [s, T)).  One wave = 32 columns (two
// 16-wide strips A/B share the streamed q A-frags -> 2x WMMA per byte).
// t-tile loop is 2x unrolled with one-tile load lookahead.
// -------------------------------------------------------------------------
__global__ __launch_bounds__(32)
void colstats_kernel(const unsigned short* __restrict__ qb,
                     const unsigned short* __restrict__ kb,
                     float* __restrict__ colmax,
                     float* __restrict__ colsum)
{
    const int lane = threadIdx.x;
    const int lrow = lane & 15;
    const int lhi  = lane >> 4;
    const int b    = blockIdx.x / (NT_TILES / 2);
    const int s0   = (blockIdx.x % (NT_TILES / 2)) * 32;

    const unsigned short* krowA = kb + ((size_t)b * T_ + s0 + lrow) * H_;
    const unsigned short* krowB = krowA + 16 * H_;
    const v16bf bkA0 = load_b_frag(krowA, lhi, 0);
    const v16bf bkA1 = load_b_frag(krowA, lhi, 32);
    const v16bf bkB0 = load_b_frag(krowB, lhi, 0);
    const v16bf bkB1 = load_b_frag(krowB, lhi, 32);

    const int sgA = s0 + lrow;         // strip-A column of this lane
    const int sgB = sgA + 16;          // strip-B column
    const unsigned short* qbase = qb + (size_t)b * T_ * H_;

    float mA = -__builtin_inff(), mB = -__builtin_inff();
    {
        const unsigned short* qr = qbase + (size_t)(s0 + lrow) * H_;
        v16bf x0 = load_a_frag(qr, lhi, 0);
        v16bf x1 = load_a_frag(qr, lhi, 32);
        for (int tt = s0; tt < T_; tt += 32) {
            const unsigned short* qr1 = qbase + (size_t)(tt + 16 + lrow) * H_;
            v16bf y0 = load_a_frag(qr1, lhi, 0);
            v16bf y1 = load_a_frag(qr1, lhi, 32);
            __builtin_prefetch(qr1 + 16 * H_, 0, 3);

            v8f aA = {}, aB = {};
            aA = WMMA_BF16(x0, bkA0, aA); aA = WMMA_BF16(x1, bkA1, aA);
            aB = WMMA_BF16(x0, bkB0, aB); aB = WMMA_BF16(x1, bkB1, aB);
            #pragma unroll
            for (int r = 0; r < 8; ++r) {
                const int tg = tt + r + lhi * 8;
                if (tg >= sgA) mA = fmaxf(mA, aA[r] * SCALE);
                if (tg >= sgB) mB = fmaxf(mB, aB[r] * SCALE);
            }

            if (tt + 32 < T_) {
                const unsigned short* qr2 = qbase + (size_t)(tt + 32 + lrow) * H_;
                x0 = load_a_frag(qr2, lhi, 0);
                x1 = load_a_frag(qr2, lhi, 32);
            }

            v8f cA = {}, cB = {};
            cA = WMMA_BF16(y0, bkA0, cA); cA = WMMA_BF16(y1, bkA1, cA);
            cB = WMMA_BF16(y0, bkB0, cB); cB = WMMA_BF16(y1, bkB1, cB);
            #pragma unroll
            for (int r = 0; r < 8; ++r) {
                const int tg = tt + 16 + r + lhi * 8;
                if (tg >= sgA) mA = fmaxf(mA, cA[r] * SCALE);
                if (tg >= sgB) mB = fmaxf(mB, cB[r] * SCALE);
            }
        }
    }
    mA = fmaxf(mA, __shfl_xor(mA, 16, 32));
    mB = fmaxf(mB, __shfl_xor(mB, 16, 32));

    float sA = 0.f, sB = 0.f;
    {
        const unsigned short* qr = qbase + (size_t)(s0 + lrow) * H_;
        v16bf x0 = load_a_frag(qr, lhi, 0);
        v16bf x1 = load_a_frag(qr, lhi, 32);
        for (int tt = s0; tt < T_; tt += 32) {
            const unsigned short* qr1 = qbase + (size_t)(tt + 16 + lrow) * H_;
            v16bf y0 = load_a_frag(qr1, lhi, 0);
            v16bf y1 = load_a_frag(qr1, lhi, 32);
            __builtin_prefetch(qr1 + 16 * H_, 0, 3);

            v8f aA = {}, aB = {};
            aA = WMMA_BF16(x0, bkA0, aA); aA = WMMA_BF16(x1, bkA1, aA);
            aB = WMMA_BF16(x0, bkB0, aB); aB = WMMA_BF16(x1, bkB1, aB);
            #pragma unroll
            for (int r = 0; r < 8; ++r) {
                const int tg = tt + r + lhi * 8;
                if (tg >= sgA) sA += __expf(aA[r] * SCALE - mA);
                if (tg >= sgB) sB += __expf(aB[r] * SCALE - mB);
            }

            if (tt + 32 < T_) {
                const unsigned short* qr2 = qbase + (size_t)(tt + 32 + lrow) * H_;
                x0 = load_a_frag(qr2, lhi, 0);
                x1 = load_a_frag(qr2, lhi, 32);
            }

            v8f cA = {}, cB = {};
            cA = WMMA_BF16(y0, bkA0, cA); cA = WMMA_BF16(y1, bkA1, cA);
            cB = WMMA_BF16(y0, bkB0, cB); cB = WMMA_BF16(y1, bkB1, cB);
            #pragma unroll
            for (int r = 0; r < 8; ++r) {
                const int tg = tt + 16 + r + lhi * 8;
                if (tg >= sgA) sA += __expf(cA[r] * SCALE - mA);
                if (tg >= sgB) sB += __expf(cB[r] * SCALE - mB);
            }
        }
    }
    sA += __shfl_xor(sA, 16, 32);
    sB += __shfl_xor(sB, 16, 32);

    if (lane < 16) {
        colmax[(size_t)b * T_ + sgA] = mA;
        colsum[(size_t)b * T_ + sgA] = sA;
        colmax[(size_t)b * T_ + sgB] = mB;
        colsum[(size_t)b * T_ + sgB] = sB;
    }
}

// -------------------------------------------------------------------------
// Kernel 3: out[t,h] = sum_s exp(aff[t,s]-colmax[s])/colsum[s] * v[s,h].
// Recompute aff via WMMA, normalize, transpose C-layout -> A-layout through
// a 16x32 bf16 LDS tile, then WMMA against transposed v.  k-frags are
// fetched one tile ahead; v-frags are hoisted to the top of each 32-key
// step so they overlap the whole stage-1 computation.
// -------------------------------------------------------------------------
__global__ __launch_bounds__(32)
void attn_out_kernel(const unsigned short* __restrict__ qb,
                     const unsigned short* __restrict__ kb,
                     const unsigned short* __restrict__ vt,
                     const float* __restrict__ colmax,
                     const float* __restrict__ colsum,
                     float* __restrict__ out)
{
    __shared__ unsigned short wtile[16 * 32];

    const int lane = threadIdx.x;
    const int lrow = lane & 15;
    const int lhi  = lane >> 4;
    const int b    = blockIdx.x / NT_TILES;
    const int t0   = (blockIdx.x % NT_TILES) * 16;

    const unsigned short* qrow = qb + ((size_t)b * T_ + (t0 + lrow)) * H_;
    const v16bf a0 = load_a_frag(qrow, lhi, 0);
    const v16bf a1 = load_a_frag(qrow, lhi, 32);

    v8f oacc[4] = { {}, {}, {}, {} };

    const int nstiles = t0 / 16 + 1;          // s-tiles with any s <= t0+15
    const unsigned short* kbase = kb + (size_t)b * T_ * H_;

    // prime k-frags for s-tile 0
    v16bf kc0 = load_b_frag(kbase + (size_t)lrow * H_, lhi, 0);
    v16bf kc1 = load_b_frag(kbase + (size_t)lrow * H_, lhi, 32);

    for (int sp = 0; sp < nstiles; sp += 2) { // 32 keys per stage-2 WMMA
        // hoist v-frags for this 32-key step (independent of stage 1)
        v16bf bv[4];
        #pragma unroll
        for (int nt = 0; nt < 4; ++nt)
            bv[nt] = load_b_frag(
                vt + ((size_t)b * H_ + nt * 16 + lrow) * T_ + sp * 16, lhi, 0);

        #pragma unroll
        for (int half = 0; half < 2; ++half) {
            const int st = sp + half;
            if (st < nstiles) {
                v8f acc = {};
                acc = WMMA_BF16(a0, kc0, acc);
                acc = WMMA_BF16(a1, kc1, acc);

                // fetch next s-tile's k-frags under the WMMAs
                if (st + 1 < nstiles) {
                    const unsigned short* kr =
                        kbase + (size_t)((st + 1) * 16 + lrow) * H_;
                    kc0 = load_b_frag(kr, lhi, 0);
                    kc1 = load_b_frag(kr, lhi, 32);
                    __builtin_prefetch(kr + 16 * H_, 0, 3);
                }

                const int   sg  = st * 16 + lrow;
                const float cm  = colmax[(size_t)b * T_ + sg];
                const float inv = 1.0f / colsum[(size_t)b * T_ + sg];
                #pragma unroll
                for (int r = 0; r < 8; ++r) {
                    const int tg = t0 + r + lhi * 8;
                    const float w =
                        (sg <= tg) ? __expf(acc[r] * SCALE - cm) * inv : 0.f;
                    wtile[(r + lhi * 8) * 32 + half * 16 + lrow] =
                        f32_to_bf16_bits(w);
                }
            } else {
                #pragma unroll
                for (int r = 0; r < 8; ++r)
                    wtile[(r + lhi * 8) * 32 + half * 16 + lrow] = 0;
            }
        }
        __syncthreads();

        // A2-frag: weight tile 16x32 in A layout (two 16B LDS vector loads)
        const unsigned short* wr = &wtile[lrow * 32];
        const v16bf aw = cat8(*(const v8bf*)(wr + lhi * 16),
                              *(const v8bf*)(wr + lhi * 16 + 8));
        __syncthreads();

        #pragma unroll
        for (int nt = 0; nt < 4; ++nt)
            oacc[nt] = WMMA_BF16(aw, bv[nt], oacc[nt]);
    }

    #pragma unroll
    for (int nt = 0; nt < 4; ++nt) {
        #pragma unroll
        for (int r = 0; r < 8; ++r) {
            out[((size_t)b * T_ + t0 + r + lhi * 8) * H_ + nt * 16 + lrow] =
                oacc[nt][r];
        }
    }
}

// -------------------------------------------------------------------------
extern "C" void kernel_launch(void* const* d_in, const int* in_sizes, int n_in,
                              void* d_out, int out_size, void* d_ws, size_t ws_size,
                              hipStream_t stream)
{
    (void)in_sizes; (void)n_in; (void)out_size; (void)ws_size;

    const float* x  = (const float*)d_in[0];
    const float* Wq = (const float*)d_in[1];
    const float* Wk = (const float*)d_in[2];
    const float* Wv = (const float*)d_in[3];
    float* out      = (float*)d_out;

    // Workspace layout (bf16 buffers as ushort)
    const size_t x_elems   = (size_t)B_ * T_ * C_;   // 16M
    const size_t w_elems   = (size_t)H_ * C_;        // 64K
    const size_t qkv_elems = (size_t)B_ * T_ * H_;   // 1M

    unsigned short* xb  = (unsigned short*)d_ws;
    unsigned short* Wqb = xb + x_elems;
    unsigned short* Wkb = Wqb + w_elems;
    unsigned short* Wvb = Wkb + w_elems;
    unsigned short* qb  = Wvb + w_elems;
    unsigned short* kb  = qb + qkv_elems;
    unsigned short* vt  = kb + qkv_elems;
    float* colmax = (float*)(vt + qkv_elems);
    float* colsum = colmax + (size_t)B_ * T_;

    // Kernel 0: bf16 conversion (x and the three weight matrices)
    {
        const int nx8 = (int)(x_elems / 8);
        cvt_bf16_kernel<<<(nx8 + 255) / 256, 256, 0, stream>>>(x, xb, nx8);
        const int nw8 = (int)(w_elems / 8);
        cvt_bf16_kernel<<<(nw8 + 255) / 256, 256, 0, stream>>>(Wq, Wqb, nw8);
        cvt_bf16_kernel<<<(nw8 + 255) / 256, 256, 0, stream>>>(Wk, Wkb, nw8);
        cvt_bf16_kernel<<<(nw8 + 255) / 256, 256, 0, stream>>>(Wv, Wvb, nw8);
    }

    const int tiles = B_ * NT_TILES;                 // 1024
    qkv_proj_kernel<<<dim3(tiles, 3), 32, 0, stream>>>(xb, Wqb, Wkb, Wvb,
                                                       qb, kb, vt);
    colstats_kernel<<<B_ * (NT_TILES / 2), 32, 0, stream>>>(qb, kb,
                                                            colmax, colsum);
    attn_out_kernel<<<tiles, 32, 0, stream>>>(qb, kb, vt, colmax, colsum, out);
}
